// ViTSelfAttention_65515431133448
// MI455X (gfx1250) — matrix operand order repaired and checked by reading, not compiled, not run on hardware
//
#include <hip/hip_runtime.h>

// ViT self-attention forward for MI455X (gfx1250, wave32, WMMA bf16 path).
// B=16, S=1024, D=768, H=12, DH=64. Compute-bound (~110 GFLOP vs ~10us of HBM
// at 23.3 TB/s; all intermediates fit in the 192MB L2), so everything routes
// through v_wmma_f32_16x16x32_bf16 with f32 accumulation.

typedef __bf16 bf16;
typedef __attribute__((ext_vector_type(16))) __bf16 v16bf;
typedef __attribute__((ext_vector_type(8)))  __bf16 v8bf;
typedef __attribute__((ext_vector_type(4)))  __bf16 v4bf;
typedef __attribute__((ext_vector_type(8)))  float  v8f;
typedef __attribute__((ext_vector_type(4)))  float  v4f;

#define S_LEN 1024
#define D_MODEL 768
#define N_HEAD 12
#define D_HEAD 64
#define M_TOT (16 * 1024)

static __device__ __forceinline__ v8f wmma_bf16(v16bf a, v16bf b, v8f c) {
  return __builtin_amdgcn_wmma_f32_16x16x32_bf16(false, a, false, b, (short)0, c, false, false);
}

// A-matrix fragment: 16 rows x 32 K. lane&15 = row; per lane K = (lane>>4)*8+[0..7]
// in elems 0..7 and 16+(lane>>4)*8+[0..7] in elems 8..15. Two b128 loads.
static __device__ __forceinline__ v16bf load_a_frag(const bf16* base, int stride, int lane) {
  const bf16* p = base + (size_t)(lane & 15) * stride + ((lane >> 4) << 3);
  v8bf lo = *(const v8bf*)(p);
  v8bf hi = *(const v8bf*)(p + 16);
  v16bf r;
#pragma unroll
  for (int i = 0; i < 8; ++i) { r[i] = lo[i]; r[i + 8] = hi[i]; }
  return r;
}

// B-matrix fragment: 32 K x 16 cols. lane&15 = column; per lane K = (lane>>4)*8+[0..7]
// in elems 0..7 and 16+(lane>>4)*8+[0..7] in elems 8..15. Two b128 loads.
static __device__ __forceinline__ v16bf load_b_frag(const bf16* base, int stride, int lane) {
  const bf16* p = base + (size_t)(lane & 15) * stride + ((lane >> 4) << 3);
  v8bf lo = *(const v8bf*)(p);
  v8bf hi = *(const v8bf*)(p + 16);
  v16bf r;
#pragma unroll
  for (int i = 0; i < 8; ++i) { r[i] = lo[i]; r[i + 8] = hi[i]; }
  return r;
}

__global__ void vit_cvt_bf16(const float* __restrict__ in, bf16* __restrict__ out, int n4) {
  int i = blockIdx.x * blockDim.x + threadIdx.x;
  if (i < n4) {
    v4f x = *(const v4f*)(in + 4 * (size_t)i);
    v4bf y;
    y[0] = (bf16)x[0]; y[1] = (bf16)x[1]; y[2] = (bf16)x[2]; y[3] = (bf16)x[3];
    *(v4bf*)(out + 4 * (size_t)i) = y;
  }
}

// Y = X @ W^T + b. One wave -> 32 rows x 64 cols (one head's features).
// z = 0:Q, 1:K (stored [B,H,S,64]), 2:V (stored transposed [B,H,64,S]).
__global__ void __launch_bounds__(128) vit_qkv_gemm(
    const bf16* __restrict__ X,
    const bf16* __restrict__ Wqb, const bf16* __restrict__ Wkb, const bf16* __restrict__ Wvb,
    const float* __restrict__ bq, const float* __restrict__ bk, const float* __restrict__ bv,
    bf16* __restrict__ Q, bf16* __restrict__ K, bf16* __restrict__ VT) {
  const int lane = threadIdx.x & 31;
  const int wave = threadIdx.x >> 5;
  const int h = blockIdx.y;
  const int z = blockIdx.z;
  const int m0 = (blockIdx.x * 4 + wave) * 32;
  const int n0 = h * D_HEAD;

  const bf16* W = (z == 0) ? Wqb : ((z == 1) ? Wkb : Wvb);
  const float* bias = (z == 0) ? bq : ((z == 1) ? bk : bv);

  v8f acc[2][4] = {};
  for (int k0 = 0; k0 < D_MODEL; k0 += 32) {
    v16bf a0 = load_a_frag(X + (size_t)m0 * D_MODEL + k0, D_MODEL, lane);
    v16bf a1 = load_a_frag(X + (size_t)(m0 + 16) * D_MODEL + k0, D_MODEL, lane);
#pragma unroll
    for (int sub = 0; sub < 4; ++sub) {
      v16bf bf = load_b_frag(W + (size_t)(n0 + sub * 16) * D_MODEL + k0, D_MODEL, lane);
      acc[0][sub] = wmma_bf16(a0, bf, acc[0][sub]);
      acc[1][sub] = wmma_bf16(a1, bf, acc[1][sub]);
    }
  }

  const int b = m0 >> 10;            // batch
  const int s0 = m0 & (S_LEN - 1);   // seq offset within batch (tiles never straddle)
  const int col = lane & 15;
  const int rh = (lane >> 4) * 8;    // C layout: row = rh + r, col = col

  if (z < 2) {
    bf16* O = (z == 0) ? Q : K;
#pragma unroll
    for (int mt = 0; mt < 2; ++mt)
#pragma unroll
      for (int sub = 0; sub < 4; ++sub) {
        float bia = bias[n0 + sub * 16 + col];
#pragma unroll
        for (int r = 0; r < 8; ++r) {
          int s = s0 + mt * 16 + rh + r;
          int d = sub * 16 + col;
          O[(((size_t)(b * N_HEAD + h)) * S_LEN + s) * D_HEAD + d] =
              (bf16)(acc[mt][sub][r] + bia);
        }
      }
  } else {
#pragma unroll
    for (int mt = 0; mt < 2; ++mt)
#pragma unroll
      for (int sub = 0; sub < 4; ++sub) {
        float bia = bias[n0 + sub * 16 + col];
        v8bf pk;
#pragma unroll
        for (int r = 0; r < 8; ++r) pk[r] = (bf16)(acc[mt][sub][r] + bia);
        // VT[b,h,d,s]: contiguous in s across the 8 accumulator rows -> b128 store
        size_t off = (((size_t)(b * N_HEAD + h)) * D_HEAD + sub * 16 + col) * S_LEN
                     + s0 + mt * 16 + rh;
        *(v8bf*)(VT + off) = pk;
      }
  }
}

// Flash attention: one wave = 16 q-rows; online softmax over k in steps of 64.
// 16 WMMAs per iteration (8 QK^T + 8 PV); reductions amortized over 4 tiles.
__global__ void __launch_bounds__(128) vit_flash_attn(
    const bf16* __restrict__ Q, const bf16* __restrict__ K,
    const bf16* __restrict__ VT, float* __restrict__ out) {
  __shared__ bf16 lds[4][16 * 64];  // per-wave 16x64 bf16 P tile (wave-private)

  const int lane = threadIdx.x & 31;
  const int wave = threadIdx.x >> 5;
  const int tile = blockIdx.x * 4 + wave;
  const int q0 = (tile & 63) * 16;   // 64 q-tiles per (b,h)
  const int bh = tile >> 6;          // 0..191

  const bf16* Qp = Q + ((size_t)bh * S_LEN + q0) * D_HEAD;
  const bf16* Kp = K + (size_t)bh * S_LEN * D_HEAD;
  const bf16* Vp = VT + (size_t)bh * D_HEAD * S_LEN;

  // Q fragments (reused across entire k loop): d 0..31 and 32..63
  v16bf aq0 = load_a_frag(Qp, D_HEAD, lane);
  v16bf aq1 = load_a_frag(Qp + 32, D_HEAD, lane);

  v8f ctx[4] = {};
  float m_run[8], l_run[8];
#pragma unroll
  for (int r = 0; r < 8; ++r) { m_run[r] = -1e30f; l_run[r] = 0.0f; }

  const float cscale = 0.125f * 1.44269504088896f;  // 1/sqrt(64) * log2(e)
  const int colc = lane & 15;
  const int rh = (lane >> 4) * 8;
  bf16* pl = lds[wave];

  for (int kc = 0; kc < S_LEN; kc += 64) {
    // ---- scores: four 16x16 tiles (k-cols kc+16j .. kc+16j+15) ----
    v8f sc[4];
#pragma unroll
    for (int j = 0; j < 4; ++j) {
      const bf16* kb = Kp + (size_t)(kc + 16 * j) * D_HEAD;
      v16bf bk0 = load_b_frag(kb, D_HEAD, lane);
      v16bf bk1 = load_b_frag(kb + 32, D_HEAD, lane);
      v8f s = {};
      s = wmma_bf16(aq0, bk0, s);
      s = wmma_bf16(aq1, bk1, s);
      sc[j] = s;
    }

    // ---- online softmax (row = rh + r; row lives in 16 lanes of same half) ----
    float corr[8];
#pragma unroll
    for (int r = 0; r < 8; ++r) {
      float x0 = sc[0][r] * cscale;
      float x1 = sc[1][r] * cscale;
      float x2 = sc[2][r] * cscale;
      float x3 = sc[3][r] * cscale;
      float t = fmaxf(fmaxf(x0, x1), fmaxf(x2, x3));
      t = fmaxf(t, __shfl_xor(t, 1));
      t = fmaxf(t, __shfl_xor(t, 2));
      t = fmaxf(t, __shfl_xor(t, 4));
      t = fmaxf(t, __shfl_xor(t, 8));
      float mn = fmaxf(m_run[r], t);
      float p0 = exp2f(x0 - mn);
      float p1 = exp2f(x1 - mn);
      float p2 = exp2f(x2 - mn);
      float p3 = exp2f(x3 - mn);
      float rs = (p0 + p1) + (p2 + p3);
      rs += __shfl_xor(rs, 1);
      rs += __shfl_xor(rs, 2);
      rs += __shfl_xor(rs, 4);
      rs += __shfl_xor(rs, 8);
      corr[r] = exp2f(m_run[r] - mn);
      l_run[r] = l_run[r] * corr[r] + rs;
      m_run[r] = mn;
      sc[0][r] = p0; sc[1][r] = p1; sc[2][r] = p2; sc[3][r] = p3;
    }
#pragma unroll
    for (int sub = 0; sub < 4; ++sub)
#pragma unroll
      for (int r = 0; r < 8; ++r) ctx[sub][r] *= corr[r];

    // ---- C-layout probs -> LDS (row-major 16x64 bf16) -> two A-layout frags ----
    // Wave-private tile + in-order LDS pipeline => no cross-wave barrier needed.
#pragma unroll
    for (int r = 0; r < 8; ++r) {
      int row = rh + r;
#pragma unroll
      for (int j = 0; j < 4; ++j)
        pl[row * 64 + 16 * j + colc] = (bf16)sc[j][r];
    }
    __builtin_amdgcn_wave_barrier();
    v16bf pf0 = load_a_frag(pl, 64, lane);        // k-cols kc   .. kc+31
    v16bf pf1 = load_a_frag(pl + 32, 64, lane);   // k-cols kc+32.. kc+63
#pragma unroll
    for (int sub = 0; sub < 4; ++sub) {
      v16bf bv0 = load_b_frag(Vp + (size_t)(sub * 16) * S_LEN + kc, S_LEN, lane);
      v16bf bv1 = load_b_frag(Vp + (size_t)(sub * 16) * S_LEN + kc + 32, S_LEN, lane);
      ctx[sub] = wmma_bf16(pf0, bv0, ctx[sub]);
      ctx[sub] = wmma_bf16(pf1, bv1, ctx[sub]);
    }
    __builtin_amdgcn_wave_barrier();
  }

  // ---- finalize: ctx / l, write [B,S,H*DH] f32 ----
  const int b = bh / N_HEAD;
  const int h = bh - b * N_HEAD;
#pragma unroll
  for (int sub = 0; sub < 4; ++sub) {
#pragma unroll
    for (int r = 0; r < 8; ++r) {
      int s = q0 + rh + r;
      out[((size_t)(b * S_LEN + s)) * D_MODEL + h * D_HEAD + sub * 16 + colc] =
          ctx[sub][r] / l_run[r];
    }
  }
}

extern "C" void kernel_launch(void* const* d_in, const int* in_sizes, int n_in,
                              void* d_out, int out_size, void* d_ws, size_t ws_size,
                              hipStream_t stream) {
  (void)in_sizes; (void)n_in; (void)out_size; (void)ws_size;
  const float* hs = (const float*)d_in[0];
  const float* Wq = (const float*)d_in[1];
  const float* bq = (const float*)d_in[2];
  const float* Wk = (const float*)d_in[3];
  const float* bk = (const float*)d_in[4];
  const float* Wv = (const float*)d_in[5];
  const float* bv = (const float*)d_in[6];
  float* out = (float*)d_out;

  const size_t szX = (size_t)M_TOT * D_MODEL;               // 12,582,912
  const size_t szW = (size_t)D_MODEL * D_MODEL;             //    589,824
  const size_t szQ = (size_t)16 * N_HEAD * S_LEN * D_HEAD;  // 12,582,912

  bf16* Xb  = (bf16*)d_ws;
  bf16* Wqb = Xb + szX;
  bf16* Wkb = Wqb + szW;
  bf16* Wvb = Wkb + szW;
  bf16* Qb  = Wvb + szW;
  bf16* Kb  = Qb + szQ;
  bf16* VTb = Kb + szQ;
  // total workspace: 2*(szX + 3*szW + 3*szQ) = ~104 MB

  vit_cvt_bf16<<<(int)(szX / 4 / 256), 256, 0, stream>>>(hs, Xb, (int)(szX / 4));
  vit_cvt_bf16<<<(int)(szW / 4 / 256), 256, 0, stream>>>(Wq, Wqb, (int)(szW / 4));
  vit_cvt_bf16<<<(int)(szW / 4 / 256), 256, 0, stream>>>(Wk, Wkb, (int)(szW / 4));
  vit_cvt_bf16<<<(int)(szW / 4 / 256), 256, 0, stream>>>(Wv, Wvb, (int)(szW / 4));

  dim3 gQKV(M_TOT / 128, N_HEAD, 3);  // 4 waves x 32 rows per block
  vit_qkv_gemm<<<gQKV, 128, 0, stream>>>(Xb, Wqb, Wkb, Wvb, bq, bk, bv, Qb, Kb, VTb);

  int tiles = 16 * N_HEAD * (S_LEN / 16);  // 12288 wave-tiles
  vit_flash_attn<<<tiles / 4, 128, 0, stream>>>(Qb, Kb, VTb, out);
}
